// AutoregressiveFlowSampler_1314259993217
// MI455X (gfx1250) — matrix operand order repaired and checked by reading, not compiled, not run on hardware
//
#include <hip/hip_runtime.h>
#include <hip/hip_bf16.h>
#include <cstdint>

// ---------------------------------------------------------------------------
// Types for CDNA5 WMMA (wave32, 16x16x32 f16 -> f32)
// ---------------------------------------------------------------------------
typedef __attribute__((ext_vector_type(16))) _Float16 v16h;
typedef __attribute__((ext_vector_type(8)))  float    v8f;

union AFrag {
    v16h v;
    _Float16 h[16];
    uint4 q[2];
};

#define NSAMP 16      // samples per block
#define SH    136     // h_in LDS row stride (halves), padded vs 128
#define SHF   520     // hff  LDS row stride (halves), padded vs 512
#define BTOT  1024    // total samples
#define NORB  32
#define NEGBIG (-1e9f)

__device__ inline v8f wmma32(v16h a, v16h b, v8f c) {
    // D = A(16x32 f16) * B(32x16 f16) + C(16x16 f32)
    return __builtin_amdgcn_wmma_f32_16x16x32_f16(
        /*neg_a=*/false, a, /*neg_b=*/false, b,
        /*c_mod=*/(short)0, c, /*reuse_a=*/false, /*reuse_b=*/false);
}

// A-fragment gather from LDS (row-major f16, stride in halves).
// ISA 16-bit A 16x32 layout: lane L -> row M=L%16; kg=L/16;
// halves j: j<8 -> K = k0..k0+7 ; j>=8 -> K = k0+16..k0+23 where k0 = ktile*32 + 8*kg.
__device__ inline v16h load_a(const _Float16* h, int stride, int ktile, int lane) {
    int m  = lane & 15;
    int kg = lane >> 4;
    int k0 = ktile * 32 + kg * 8;
    AFrag f;
    f.q[0] = *(const uint4*)(h + m * stride + k0);
    f.q[1] = *(const uint4*)(h + m * stride + k0 + 16);
    return f.v;
}

// B-fragment: pre-packed contiguous 16 halves per lane.
__device__ inline v16h load_b(const _Float16* wp, int tile, int lane) {
    const _Float16* p = wp + ((size_t)tile * 32 + lane) * 16;
    AFrag f;
    f.q[0] = *(const uint4*)(p);
    f.q[1] = *(const uint4*)(p + 8);
    return f.v;
}

__device__ inline void ln_row(const float* __restrict__ x, _Float16* __restrict__ out,
                              const float* __restrict__ w, const float* __restrict__ b) {
    float m = 0.f;
    #pragma unroll 8
    for (int d = 0; d < 128; ++d) m += x[d];
    m *= (1.f / 128.f);
    float v = 0.f;
    #pragma unroll 8
    for (int d = 0; d < 128; ++d) { float dd = x[d] - m; v += dd * dd; }
    v *= (1.f / 128.f);
    float rs = rsqrtf(v + 1e-5f);
    #pragma unroll 4
    for (int d = 0; d < 128; ++d)
        out[d] = (_Float16)((x[d] - m) * rs * w[d] + b[d]);
}

__device__ inline float gelu_exact(float x) {
    return 0.5f * x * (1.f + erff(x * 0.70710678118f));
}

__device__ inline float gumbel_from_id(uint32_t id) {
    uint32_t x = id * 0x9E3779B9u + 0x85EBCA6Bu;
    x ^= x >> 16; x *= 0x7FEB352Du;
    x ^= x >> 15; x *= 0x846CA68Bu;
    x ^= x >> 16;
    float u = (float)(x >> 8) * (1.0f / 16777216.0f);
    u = fminf(fmaxf(u, 1e-7f), 1.0f - 1e-7f);
    return -logf(-logf(u));
}

// ---------------------------------------------------------------------------
// Weight packing: W (Nfeat x K, row-major f32) -> per-tile WMMA B fragments f16.
// dst element idx = l*perLayer + ((nt*KT + kt)*32 + lane)*16 + j
// holds B[k][n] = W[n][k], n = nt*16 + lane%16, k = kt*32 + (lane/16)*8 + (j<8 ? j : j+8).
// ---------------------------------------------------------------------------
__global__ __launch_bounds__(256) void pack_kernel(const float* __restrict__ W,
                                                   _Float16* __restrict__ out,
                                                   int Nt, int KT, int layers) {
    int idx = blockIdx.x * 256 + threadIdx.x;
    int perLayer = Nt * KT * 512;
    int total = layers * perLayer;
    if (idx >= total) return;
    int l = idx / perLayer;
    int r = idx - l * perLayer;
    int tile = r >> 9;
    int w    = r & 511;
    int lane = w >> 4;
    int j    = w & 15;
    int nt = tile / KT;
    int kt = tile - nt * KT;
    int n = nt * 16 + (lane & 15);
    int k = kt * 32 + (lane >> 4) * 8 + (j < 8 ? j : j + 8);
    int K  = KT * 32;
    int Nf = Nt * 16;
    out[idx] = (_Float16)W[(size_t)l * Nf * K + (size_t)n * K + k];
}

__global__ __launch_bounds__(256) void init_kernel(int* tok, int* arem, int* brem, float* logp) {
    int b = blockIdx.x * 256 + threadIdx.x;
    if (b < BTOT) { tok[b] = 4; arem[b] = 16; brem[b] = 16; logp[b] = 0.f; }
}

__global__ __launch_bounds__(256) void embed_kernel(float* __restrict__ xg,
                                                    const float* __restrict__ se,
                                                    const float* __restrict__ pe,
                                                    const int* __restrict__ tok, int t) {
    int i = blockIdx.x * 256 + threadIdx.x;   // 0 .. 1024*128-1
    int b = i >> 7, d = i & 127;
    xg[i] = se[tok[b] * 128 + d] + pe[t * 128 + d];
}

// ---------------------------------------------------------------------------
// Fused transformer layer (one kernel per (step t, layer l)).
// Block: 256 threads (8 wave32), 16 samples. Weights pre-packed f16 fragments.
// ---------------------------------------------------------------------------
__global__ __launch_bounds__(256) void layer_kernel(
    float* __restrict__ xg,
    const float* __restrict__ ln1w, const float* __restrict__ ln1b,
    const float* __restrict__ inb,  const float* __restrict__ outb,
    const float* __restrict__ ln2w, const float* __restrict__ ln2b,
    const float* __restrict__ fb1,  const float* __restrict__ fb2,
    const _Float16* __restrict__ winp, const _Float16* __restrict__ woutp,
    const _Float16* __restrict__ w1p,  const _Float16* __restrict__ w2p,
    _Float16* __restrict__ kcache, _Float16* __restrict__ vcache, int t)
{
    __shared__ __align__(16) _Float16 h_in[NSAMP * SH];     // LN out / attn out / LN2 out (f16 A-matrix)
    __shared__ __align__(16) float    xs[NSAMP * 128];      // residual stream (f32)
    __shared__ __align__(16) float    qs[NSAMP * 128];      // q (f32)
    __shared__ __align__(16) _Float16 kt_s[NSAMP * 128];    // K at pos t
    __shared__ __align__(16) _Float16 vt_s[NSAMP * 128];    // V at pos t
    __shared__ __align__(16) _Float16 hff[NSAMP * SHF];     // gelu(W1 h2) (f16 A-matrix)

    const int tid  = threadIdx.x;
    const int lane = tid & 31;
    // wave index is wave-invariant under wave32: force into an SGPR so tile
    // selection / category branches become scalar branches, not exec-mask ops.
    const int wvu  = __builtin_amdgcn_readfirstlane(tid >> 5);
    const int kg   = lane >> 4;
    const int b0   = blockIdx.x * NSAMP;

    // ---- load residual tile ----
    for (int i = tid; i < NSAMP * 128; i += 256) xs[i] = xg[b0 * 128 + i];
    __syncthreads();

    // ---- LN1 ----
    if (tid < NSAMP) ln_row(xs + tid * 128, h_in + tid * SH, ln1w, ln1b);
    __syncthreads();

    // ---- QKV: 24 N-tiles, 3 per wave, K = 128 (4 k-tiles) ----
    {
        v16h a0 = load_a(h_in, SH, 0, lane);
        v16h a1 = load_a(h_in, SH, 1, lane);
        v16h a2 = load_a(h_in, SH, 2, lane);
        v16h a3 = load_a(h_in, SH, 3, lane);
        for (int i = 0; i < 3; ++i) {
            const int nt = wvu * 3 + i;              // uniform (SGPR)
            const int ncol = nt * 16 + (lane & 15);
            float bv = inb[ncol];
            v8f acc;
            #pragma unroll
            for (int r = 0; r < 8; ++r) acc[r] = bv;
            acc = wmma32(a0, load_b(winp, nt * 4 + 0, lane), acc);
            acc = wmma32(a1, load_b(winp, nt * 4 + 1, lane), acc);
            acc = wmma32(a2, load_b(winp, nt * 4 + 2, lane), acc);
            acc = wmma32(a3, load_b(winp, nt * 4 + 3, lane), acc);
            // Uniform branch on tile category (16-col aligned splits).
            if (nt < 8) {                            // Q -> LDS f32
                #pragma unroll
                for (int r = 0; r < 8; ++r)
                    qs[(kg * 8 + r) * 128 + ncol] = acc[r];
            } else if (nt < 16) {                    // K -> LDS f16 + global cache
                const int c = ncol - 128;
                const int hh = c >> 5, hd = c & 31;
                #pragma unroll
                for (int r = 0; r < 8; ++r) {
                    int m = kg * 8 + r;
                    _Float16 hv = (_Float16)acc[r];
                    kt_s[m * 128 + c] = hv;
                    kcache[((b0 + m) * 4 + hh) * (NORB * 32) + t * 32 + hd] = hv;
                }
            } else {                                 // V -> LDS f16 + global cache
                const int c = ncol - 256;
                const int hh = c >> 5, hd = c & 31;
                #pragma unroll
                for (int r = 0; r < 8; ++r) {
                    int m = kg * 8 + r;
                    _Float16 hv = (_Float16)acc[r];
                    vt_s[m * 128 + c] = hv;
                    vcache[((b0 + m) * 4 + hh) * (NORB * 32) + t * 32 + hd] = hv;
                }
            }
        }
    }
    __syncthreads();

    // ---- attention: one thread per (sample, head), online softmax over p<=t ----
    if (tid < NSAMP * 4) {
        int m = tid >> 2, hh = tid & 3;
        float qv[32];
        #pragma unroll
        for (int d = 0; d < 32; ++d) qv[d] = qs[m * 128 + hh * 32 + d];
        float att[32];
        #pragma unroll
        for (int d = 0; d < 32; ++d) att[d] = 0.f;
        float M = -1e30f, S = 0.f;
        const float scale = 0.1767766953f; // 1/sqrt(32)
        for (int p = 0; p <= t; ++p) {
            const _Float16* kp;
            const _Float16* vp;
            if (p < t) {
                int base = ((b0 + m) * 4 + hh) * (NORB * 32) + p * 32;
                kp = kcache + base;
                vp = vcache + base;
            } else {
                kp = kt_s + m * 128 + hh * 32;
                vp = vt_s + m * 128 + hh * 32;
            }
            float s = 0.f;
            #pragma unroll
            for (int d = 0; d < 32; ++d) s += qv[d] * (float)kp[d];
            s *= scale;
            float nm = fmaxf(M, s);
            float eo = __expf(M - nm), en = __expf(s - nm);
            S = S * eo + en;
            #pragma unroll
            for (int d = 0; d < 32; ++d) att[d] = att[d] * eo + en * (float)vp[d];
            M = nm;
        }
        float inv = 1.f / S;
        #pragma unroll
        for (int d = 0; d < 32; ++d)
            h_in[m * SH + hh * 32 + d] = (_Float16)(att[d] * inv);
    }
    __syncthreads();

    // ---- out_proj: 8 N-tiles, 1 per wave; residual add into xs ----
    {
        v16h a0 = load_a(h_in, SH, 0, lane);
        v16h a1 = load_a(h_in, SH, 1, lane);
        v16h a2 = load_a(h_in, SH, 2, lane);
        v16h a3 = load_a(h_in, SH, 3, lane);
        const int nt = wvu;                          // uniform
        const int ncol = nt * 16 + (lane & 15);
        float bv = outb[ncol];
        v8f acc;
        #pragma unroll
        for (int r = 0; r < 8; ++r) acc[r] = bv;
        acc = wmma32(a0, load_b(woutp, nt * 4 + 0, lane), acc);
        acc = wmma32(a1, load_b(woutp, nt * 4 + 1, lane), acc);
        acc = wmma32(a2, load_b(woutp, nt * 4 + 2, lane), acc);
        acc = wmma32(a3, load_b(woutp, nt * 4 + 3, lane), acc);
        #pragma unroll
        for (int r = 0; r < 8; ++r) xs[(kg * 8 + r) * 128 + ncol] += acc[r];
    }
    __syncthreads();

    // ---- LN2 ----
    if (tid < NSAMP) ln_row(xs + tid * 128, h_in + tid * SH, ln2w, ln2b);
    __syncthreads();

    // ---- FFN W1 + exact GELU: 32 N-tiles, 4 per wave ----
    {
        v16h a0 = load_a(h_in, SH, 0, lane);
        v16h a1 = load_a(h_in, SH, 1, lane);
        v16h a2 = load_a(h_in, SH, 2, lane);
        v16h a3 = load_a(h_in, SH, 3, lane);
        for (int i = 0; i < 4; ++i) {
            const int nt = wvu * 4 + i;              // uniform
            const int ncol = nt * 16 + (lane & 15);
            float bv = fb1[ncol];
            v8f acc;
            #pragma unroll
            for (int r = 0; r < 8; ++r) acc[r] = bv;
            acc = wmma32(a0, load_b(w1p, nt * 4 + 0, lane), acc);
            acc = wmma32(a1, load_b(w1p, nt * 4 + 1, lane), acc);
            acc = wmma32(a2, load_b(w1p, nt * 4 + 2, lane), acc);
            acc = wmma32(a3, load_b(w1p, nt * 4 + 3, lane), acc);
            #pragma unroll
            for (int r = 0; r < 8; ++r)
                hff[(kg * 8 + r) * SHF + ncol] = (_Float16)gelu_exact(acc[r]);
        }
    }
    __syncthreads();

    // ---- FFN W2: 8 N-tiles, 1 per wave, K = 512 (16 k-tiles); residual add ----
    {
        const int nt = wvu;                          // uniform
        const int ncol = nt * 16 + (lane & 15);
        float bv = fb2[ncol];
        v8f acc;
        #pragma unroll
        for (int r = 0; r < 8; ++r) acc[r] = bv;
        for (int kt = 0; kt < 16; ++kt) {
            v16h a = load_a(hff, SHF, kt, lane);
            acc = wmma32(a, load_b(w2p, nt * 16 + kt, lane), acc);
        }
        #pragma unroll
        for (int r = 0; r < 8; ++r) xs[(kg * 8 + r) * 128 + ncol] += acc[r];
    }
    __syncthreads();

    // ---- write back residual ----
    for (int i = tid; i < NSAMP * 128; i += 256) xg[b0 * 128 + i] = xs[i];
}

// ---------------------------------------------------------------------------
// Final LN + head + validity mask + Gumbel-max sample + state update.
// ---------------------------------------------------------------------------
__global__ __launch_bounds__(256) void head_kernel(
    const float* __restrict__ xg,
    const float* __restrict__ fnw, const float* __restrict__ fnb,
    const float* __restrict__ hw,  const float* __restrict__ hb,
    int* __restrict__ tok, int* __restrict__ arem, int* __restrict__ brem,
    float* __restrict__ logp, float* __restrict__ dout, int t)
{
    int b = blockIdx.x * 256 + threadIdx.x;
    const float* x = xg + b * 128;

    float m = 0.f;
    #pragma unroll 8
    for (int d = 0; d < 128; ++d) m += x[d];
    m *= (1.f / 128.f);
    float v = 0.f;
    #pragma unroll 8
    for (int d = 0; d < 128; ++d) { float dd = x[d] - m; v += dd * dd; }
    v *= (1.f / 128.f);
    float rs = rsqrtf(v + 1e-5f);

    float lg[4] = { hb[0], hb[1], hb[2], hb[3] };
    #pragma unroll 4
    for (int d = 0; d < 128; ++d) {
        float xf = (x[d] - m) * rs * fnw[d] + fnb[d];
        lg[0] += xf * hw[0 * 128 + d];
        lg[1] += xf * hw[1 * 128 + d];
        lg[2] += xf * hw[2 * 128 + d];
        lg[3] += xf * hw[3 * 128 + d];
    }

    int ar = arem[b], br = brem[b];
    int after = NORB - 1 - t;
    const int AST[4] = {0, 0, 1, 1};
    const int BST[4] = {0, 1, 0, 1};
    float ml[4];
    #pragma unroll
    for (int j = 0; j < 4; ++j) {
        bool ok = (ar >= AST[j]) && (br >= BST[j]) &&
                  (ar - AST[j] <= after) && (br - BST[j] <= after);
        ml[j] = ok ? lg[j] : NEGBIG;
    }
    float mx = fmaxf(fmaxf(ml[0], ml[1]), fmaxf(ml[2], ml[3]));
    float sum = __expf(ml[0] - mx) + __expf(ml[1] - mx) +
                __expf(ml[2] - mx) + __expf(ml[3] - mx);
    float lse = logf(sum);

    int s = 0;
    float best = -1e30f;
    #pragma unroll
    for (int j = 0; j < 4; ++j) {
        uint32_t id = ((uint32_t)(t * BTOT + b)) * 4u + (uint32_t)j + 42u;
        float cand = ml[j] + gumbel_from_id(id);
        if (cand > best) { best = cand; s = j; }
    }

    float lp = logp[b] + (ml[s] - mx - lse);
    logp[b] = lp;
    arem[b] = ar - (s >> 1);
    brem[b] = br - (s & 1);
    tok[b]  = s;

    // configs: (B, 64) = [alpha(32) | beta(32)] then logp (B)
    dout[b * 64 + t]       = (float)(s >> 1);
    dout[b * 64 + 32 + t]  = (float)(s & 1);
    if (t == NORB - 1) dout[BTOT * 64 + b] = lp;
}

// ---------------------------------------------------------------------------
// Host launcher
// ---------------------------------------------------------------------------
extern "C" void kernel_launch(void* const* d_in, const int* in_sizes, int n_in,
                              void* d_out, int out_size, void* d_ws, size_t ws_size,
                              hipStream_t stream) {
    (void)in_sizes; (void)n_in; (void)out_size; (void)ws_size;

    const float* state_emb = (const float*)d_in[1];
    const float* pos_emb   = (const float*)d_in[2];
    const float* ln1w = (const float*)d_in[3];
    const float* ln1b = (const float*)d_in[4];
    const float* inw  = (const float*)d_in[5];
    const float* inb  = (const float*)d_in[6];
    const float* outw = (const float*)d_in[7];
    const float* outb = (const float*)d_in[8];
    const float* ln2w = (const float*)d_in[9];
    const float* ln2b = (const float*)d_in[10];
    const float* w1   = (const float*)d_in[11];
    const float* b1   = (const float*)d_in[12];
    const float* w2   = (const float*)d_in[13];
    const float* b2   = (const float*)d_in[14];
    const float* fnw  = (const float*)d_in[15];
    const float* fnb  = (const float*)d_in[16];
    const float* hw   = (const float*)d_in[17];
    const float* hb   = (const float*)d_in[18];

    char* ws = (char*)d_ws;
    size_t off = 0;
    auto take = [&](size_t bytes) -> char* {
        char* p = ws + off;
        off += (bytes + 255) & ~(size_t)255;
        return p;
    };
    float*    xg   = (float*)   take((size_t)BTOT * 128 * 4);
    int*      tok  = (int*)     take((size_t)BTOT * 4);
    int*      arem = (int*)     take((size_t)BTOT * 4);
    int*      brem = (int*)     take((size_t)BTOT * 4);
    float*    logp = (float*)   take((size_t)BTOT * 4);
    _Float16* kc   = (_Float16*)take((size_t)4 * BTOT * 4 * NORB * 32 * 2);
    _Float16* vc   = (_Float16*)take((size_t)4 * BTOT * 4 * NORB * 32 * 2);
    _Float16* winp = (_Float16*)take((size_t)4 * 24 * 4 * 512 * 2);
    _Float16* wop  = (_Float16*)take((size_t)4 *  8 * 4 * 512 * 2);
    _Float16* w1p  = (_Float16*)take((size_t)4 * 32 * 4 * 512 * 2);
    _Float16* w2p  = (_Float16*)take((size_t)4 *  8 * 16 * 512 * 2);

    const size_t kcLayerStride = (size_t)BTOT * 4 * NORB * 32;

    init_kernel<<<4, 256, 0, stream>>>(tok, arem, brem, logp);
    pack_kernel<<<(4 * 24 * 4 * 512 + 255) / 256, 256, 0, stream>>>(inw,  winp, 24, 4, 4);
    pack_kernel<<<(4 *  8 * 4 * 512 + 255) / 256, 256, 0, stream>>>(outw, wop,   8, 4, 4);
    pack_kernel<<<(4 * 32 * 4 * 512 + 255) / 256, 256, 0, stream>>>(w1,   w1p,  32, 4, 4);
    pack_kernel<<<(4 *  8 * 16 * 512 + 255) / 256, 256, 0, stream>>>(w2,  w2p,   8, 16, 4);

    for (int t = 0; t < NORB; ++t) {
        embed_kernel<<<(BTOT * 128) / 256, 256, 0, stream>>>(xg, state_emb, pos_emb, tok, t);
        for (int l = 0; l < 4; ++l) {
            layer_kernel<<<BTOT / NSAMP, 256, 0, stream>>>(
                xg,
                ln1w + l * 128, ln1b + l * 128, inb + l * 384, outb + l * 128,
                ln2w + l * 128, ln2b + l * 128, b1 + l * 512, b2 + l * 128,
                winp + (size_t)l * 24 * 4 * 512,
                wop  + (size_t)l *  8 * 4 * 512,
                w1p  + (size_t)l * 32 * 4 * 512,
                w2p  + (size_t)l *  8 * 16 * 512,
                kc + l * kcLayerStride, vc + l * kcLayerStride, t);
        }
        head_kernel<<<BTOT / 256, 256, 0, stream>>>(
            xg, fnw, fnb, hw, hb, tok, arem, brem, logp, (float*)d_out, t);
    }
}
